// GCN_55207509623327
// MI455X (gfx1250) — matrix operand order repaired and checked by reading, not compiled, not run on hardware
//
#include <hip/hip_runtime.h>
#include <hip/hip_bf16.h>

typedef __attribute__((ext_vector_type(16))) __bf16 v16bf;
typedef __attribute__((ext_vector_type(8)))  __bf16 v8bf;
typedef __attribute__((ext_vector_type(8)))  float  v8f;
typedef int v4i __attribute__((vector_size(16)));

// ---- CDNA5 async global->LDS copy (guarded; falls back to load+ds_store) ----
#if defined(__has_builtin)
#  if __has_builtin(__builtin_amdgcn_global_load_async_to_lds_b128)
#    define USE_ASYNC_LDS 1
#  endif
#endif
#ifndef USE_ASYNC_LDS
#  define USE_ASYNC_LDS 0
#endif

#if USE_ASYNC_LDS
// signature (from clang diagnostic): (v4i AS(1)* gsrc, v4i AS(3)* ldst, imm offset, imm cpol)
#define ASYNC_CP16(gp, lp, off)                                          \
    __builtin_amdgcn_global_load_async_to_lds_b128(                      \
        (__attribute__((address_space(1))) v4i*)(gp),                    \
        (__attribute__((address_space(3))) v4i*)(lp), (off), 0)
__device__ __forceinline__ void wait_async0() {
#if __has_builtin(__builtin_amdgcn_s_wait_asynccnt)
    __builtin_amdgcn_s_wait_asynccnt(0);
#else
    asm volatile("s_wait_asynccnt 0" ::: "memory");
#endif
}
#endif

// ---------------- elementwise / graph kernels ----------------

__global__ void k_fill_f32(float* p, float v, long n) {
    long i = (long)blockIdx.x * blockDim.x + threadIdx.x;
    if (i < n) p[i] = v;
}

__global__ void k_count_deg(const int* __restrict__ dst, float* __restrict__ deg, int E) {
    int e = blockIdx.x * blockDim.x + threadIdx.x;
    if (e < E)
        __hip_atomic_fetch_add(&deg[dst[e]], 1.0f, __ATOMIC_RELAXED, __HIP_MEMORY_SCOPE_AGENT);
}

__global__ void k_rsqrt_inplace(float* p, int n) {
    int i = blockIdx.x * blockDim.x + threadIdx.x;
    if (i < n) p[i] = rsqrtf(p[i]);
}

__global__ void k_cvt_bf16(const float* __restrict__ in, __bf16* __restrict__ out, long n) {
    long i = (long)blockIdx.x * blockDim.x + threadIdx.x;
    if (i < n) out[i] = (__bf16)in[i];
}

// out[n,f] = h[n,f] * dinv[n]^2   (self-loop term; also initializes the accumulator)
__global__ void k_agg_self(const float* __restrict__ h, const float* __restrict__ dinv,
                           float* __restrict__ out, int N, int F) {
    long i = (long)blockIdx.x * blockDim.x + threadIdx.x;
    if (i < (long)N * F) {
        int n = (int)(i / F);
        float w = dinv[n];
        out[i] = h[i] * (w * w);
    }
}

// one wave32 per edge: out[dst] += h[src] * dinv[src]*dinv[dst]
__global__ void k_agg_edges(const int* __restrict__ src, const int* __restrict__ dst,
                            const float* __restrict__ dinv, const float* __restrict__ h,
                            float* __restrict__ out, int E, int F) {
    long gt = (long)blockIdx.x * blockDim.x + threadIdx.x;
    int gw = (int)(gt >> 5);
    if (gw >= E) return;
    int lane = (int)(gt & 31);
    int s = src[gw], d = dst[gw];
    float w = dinv[s] * dinv[d];
    const float* hs = h + (size_t)s * F;
    float* od = out + (size_t)d * F;
    for (int f = lane; f < F; f += 32)
        __hip_atomic_fetch_add(&od[f], hs[f] * w, __ATOMIC_RELAXED, __HIP_MEMORY_SCOPE_AGENT);
}

// out_bf16[n,f] = act(agg[n,f] + bias[f]); leaky=0 -> ReLU, leaky=1 -> LeakyReLU(0.01)
__global__ void k_bias_act_cvt(const float* __restrict__ agg, const float* __restrict__ bias,
                               __bf16* __restrict__ out, long total, int F, int leaky) {
    long i = (long)blockIdx.x * blockDim.x + threadIdx.x;
    if (i < total) {
        int f = (int)(i % F);
        float v = agg[i] + bias[f];
        float neg = leaky ? 0.01f : 0.0f;
        v = (v >= 0.0f) ? v : neg * v;
        out[i] = (__bf16)v;
    }
}

// ---------------- bf16 WMMA GEMM:  C[M,Nn] = A[M,K] * B[K,Nn] (+bias) ----------------
// Block tile 128x64, 8 waves, 4 accumulators/wave, K stepped by 32.
// Double-buffered LDS pipeline: stage tile t+1 (async-to-LDS when available)
// while computing tile t; one barrier per K-step.

#define BM 128
#define BN 64
#define BK 32
#define LDT 40   // padded K-stride in LDS (80B rows -> 16B aligned, conflict-spread)

__launch_bounds__(256)
__global__ void k_gemm_bf16(const __bf16* __restrict__ A, const __bf16* __restrict__ B,
                            float* __restrict__ C, const float* __restrict__ bias,
                            int M, int Nn, int K) {
    __shared__ __align__(16) __bf16 As[2][BM][LDT];
    __shared__ __align__(16) __bf16 Bs[2][BN][LDT];   // transposed: Bs[b][n][k]

    const int tid  = threadIdx.x;
    const int lane = tid & 31;
    const int wv   = tid >> 5;           // 0..7
    const int hi   = lane >> 4;          // lane half
    const int lm   = lane & 15;
    const int m0   = blockIdx.y * BM;
    const int n0   = blockIdx.x * BN;

    v8f acc[4] = {};

    // A staging map: thread t -> row t>>1, 16-element half (t&1)*16
    const int arow  = tid >> 1;
    const int ahalf = (tid & 1) * 16;
    int grA = m0 + arow;
    if (grA >= M) grA = M - 1;           // clamp: rows >= M are computed but masked at store
    const __bf16* aRow = A + (size_t)grA * K + ahalf;

    // B staging map (vector path): thread t -> k row t>>3, 8-col segment (t&7)*8
    const int bkk = tid >> 3;            // 0..31
    const int bns = (tid & 7) * 8;       // 0..56
    const bool fullN = (n0 + BN <= Nn) && ((Nn & 7) == 0);
    const __bf16* bRowV = B + (size_t)bkk * Nn + n0 + bns;

    auto stage = [&](int sb, int kk0) {
        // ---- A tile: two b128 copies per thread ----
        const __bf16* ap = aRow + kk0;
#if USE_ASYNC_LDS
        ASYNC_CP16(ap, &As[sb][arow][ahalf], 0);
        ASYNC_CP16(ap, &As[sb][arow][ahalf], 16);  // imm offset applies to both addrs
#else
        v8bf a0 = *(const v8bf*)ap;
        v8bf a1 = *(const v8bf*)(ap + 8);
        if (kk0 + BK < K) __builtin_prefetch(ap + BK, 0, 1);  // global_prefetch_b8
        *(v8bf*)&As[sb][arow][ahalf]     = a0;
        *(v8bf*)&As[sb][arow][ahalf + 8] = a1;
#endif
        // ---- B tile, transposed into LDS ----
        if (fullN) {
            v8bf bv = *(const v8bf*)(bRowV + (size_t)kk0 * Nn);
            #pragma unroll
            for (int j = 0; j < 8; ++j) Bs[sb][bns + j][bkk] = bv[j];
        } else {
            for (int idx = tid; idx < BK * BN; idx += 256) {
                int kk = idx >> 6;               // BN == 64
                int nn = idx & (BN - 1);
                int gn = n0 + nn;
                __bf16 v = (__bf16)0.0f;
                if (gn < Nn) v = B[(size_t)(kk0 + kk) * Nn + gn];
                Bs[sb][nn][kk] = v;
            }
        }
    };

    auto pipe_barrier = [&]() {
#if USE_ASYNC_LDS
        wait_async0();                   // own async copies landed in LDS
#endif
        __syncthreads();                 // everyone's copies visible / reads retired
    };

    auto compute = [&](int cb) {
        // A fragment: lane(m=lm, hi) holds A[m][hi*8..+7] and A[m][16+hi*8..+7]
        const int mrow = wv * 16 + lm;
        v8bf alo = *(const v8bf*)&As[cb][mrow][hi * 8];
        v8bf ahv = *(const v8bf*)&As[cb][mrow][16 + hi * 8];
        v16bf afrag = __builtin_shufflevector(alo, ahv,
            0, 1, 2, 3, 4, 5, 6, 7, 8, 9, 10, 11, 12, 13, 14, 15);
        // Preload all B fragments, then issue WMMAs back-to-back
        v16bf bfrag[4];
        #pragma unroll
        for (int c = 0; c < 4; ++c) {
            const int nc = c * 16 + lm;
            v8bf b0 = *(const v8bf*)&Bs[cb][nc][hi * 8];
            v8bf b1 = *(const v8bf*)&Bs[cb][nc][16 + hi * 8];
            bfrag[c] = __builtin_shufflevector(b0, b1,
                0, 1, 2, 3, 4, 5, 6, 7, 8, 9, 10, 11, 12, 13, 14, 15);
        }
        #pragma unroll
        for (int c = 0; c < 4; ++c)
            acc[c] = __builtin_amdgcn_wmma_f32_16x16x32_bf16(
                false, afrag, false, bfrag[c], (short)0, acc[c], false, false);
    };

    // ---- software pipeline ----
    stage(0, 0);
    pipe_barrier();
    int k0 = 0, buf = 0;
    #pragma unroll 1
    for (;;) {
        if (k0 + BK < K) stage(buf ^ 1, k0 + BK);
        compute(buf);
        pipe_barrier();
        k0 += BK;
        if (k0 >= K) break;
        buf ^= 1;
    }

    // ---- store: D VGPR r -> row (wv*16 + hi*8 + r), col (c*16 + lm) ----
    #pragma unroll
    for (int c = 0; c < 4; ++c) {
        int gcol = n0 + c * 16 + lm;
        if (gcol >= Nn) continue;
        float bv = bias ? bias[gcol] : 0.0f;
        #pragma unroll
        for (int r = 0; r < 8; ++r) {
            int gr = m0 + wv * 16 + hi * 8 + r;
            if (gr < M) C[(size_t)gr * Nn + gcol] = acc[c][r] + bv;
        }
    }
}

// ---------------- host-side orchestration ----------------

static inline unsigned ceil_div_u(long a, long b) { return (unsigned)((a + b - 1) / b); }

extern "C" void kernel_launch(void* const* d_in, const int* in_sizes, int n_in,
                              void* d_out, int out_size, void* d_ws, size_t ws_size,
                              hipStream_t stream) {
    (void)n_in; (void)out_size; (void)ws_size;
    const float* x  = (const float*)d_in[0];
    const int*   ei = (const int*)  d_in[1];
    const float* W1 = (const float*)d_in[2];
    const float* b1 = (const float*)d_in[3];
    const float* W2 = (const float*)d_in[4];
    const float* b2 = (const float*)d_in[5];
    const float* Wm = (const float*)d_in[6];
    const float* bm = (const float*)d_in[7];
    float* out = (float*)d_out;

    const int F1  = in_sizes[3];            // 256
    const int F2  = in_sizes[5];            // 128
    const int FO  = in_sizes[7];            // 7
    const int FIN = in_sizes[2] / F1;       // 512
    const int N   = in_sizes[0] / FIN;      // 50000
    const int E   = in_sizes[1] / 2;        // 800000
    const int* src = ei;
    const int* dst = ei + E;

    // workspace layout (regions reused across phases)
    size_t off = 0;
    auto alloc = [&](size_t bytes) -> char* {
        char* p = (char*)d_ws + off;
        off += (bytes + 255) & ~(size_t)255;
        return p;
    };
    const size_t Np = (size_t)((N + 127) & ~127);
    __bf16* xbf  = (__bf16*)alloc(Np * FIN * 2);          // later reused for h1bf / h2bf
    __bf16* w1b  = (__bf16*)alloc((size_t)FIN * F1 * 2);
    __bf16* w2b  = (__bf16*)alloc((size_t)F1 * F2 * 2);
    __bf16* wmb  = (__bf16*)alloc((size_t)F2 * FO * 2);
    float*  dinv = (float*)alloc(Np * 4);
    float*  bufP = (float*)alloc(Np * F1 * 4);            // h1lin, later h2lin
    float*  bufQ = (float*)alloc(Np * F1 * 4);            // h1agg, later h2agg

    const unsigned T = 256;

    // degree (self-loop => init 1.0) -> dinv = rsqrt(deg)
    k_fill_f32<<<ceil_div_u(N, T), T, 0, stream>>>(dinv, 1.0f, N);
    k_count_deg<<<ceil_div_u(E, T), T, 0, stream>>>(dst, dinv, E);
    k_rsqrt_inplace<<<ceil_div_u(N, T), T, 0, stream>>>(dinv, N);

    // fp32 -> bf16 conversions
    k_cvt_bf16<<<ceil_div_u((long)N * FIN, T), T, 0, stream>>>(x,  xbf, (long)N * FIN);
    k_cvt_bf16<<<ceil_div_u((long)FIN * F1, T), T, 0, stream>>>(W1, w1b, (long)FIN * F1);
    k_cvt_bf16<<<ceil_div_u((long)F1 * F2, T), T, 0, stream>>>(W2, w2b, (long)F1 * F2);
    k_cvt_bf16<<<ceil_div_u((long)F2 * FO, T), T, 0, stream>>>(Wm, wmb, (long)F2 * FO);

    // layer 1: h1lin = xbf @ W1
    dim3 g1(ceil_div_u(F1, BN), ceil_div_u(N, BM));
    k_gemm_bf16<<<g1, T, 0, stream>>>(xbf, w1b, bufP, nullptr, N, F1, FIN);
    // aggregate (self-loops initialize, edges atomically accumulate)
    k_agg_self<<<ceil_div_u((long)N * F1, T), T, 0, stream>>>(bufP, dinv, bufQ, N, F1);
    k_agg_edges<<<ceil_div_u((long)E * 32, T), T, 0, stream>>>(src, dst, dinv, bufP, bufQ, E, F1);
    // + b1, ReLU, -> bf16 (reuse xbf region; x is dead)
    __bf16* h1bf = xbf;
    k_bias_act_cvt<<<ceil_div_u((long)N * F1, T), T, 0, stream>>>(bufQ, b1, h1bf, (long)N * F1, F1, 0);

    // layer 2: h2lin = h1bf @ W2
    dim3 g2(ceil_div_u(F2, BN), ceil_div_u(N, BM));
    k_gemm_bf16<<<g2, T, 0, stream>>>(h1bf, w2b, bufP, nullptr, N, F2, F1);
    k_agg_self<<<ceil_div_u((long)N * F2, T), T, 0, stream>>>(bufP, dinv, bufQ, N, F2);
    k_agg_edges<<<ceil_div_u((long)E * 32, T), T, 0, stream>>>(src, dst, dinv, bufP, bufQ, E, F2);
    // + b2, LeakyReLU(0.01), -> bf16 (reuse xbf region again; h1bf is dead)
    __bf16* h2bf = xbf;
    k_bias_act_cvt<<<ceil_div_u((long)N * F2, T), T, 0, stream>>>(bufQ, b2, h2bf, (long)N * F2, F2, 1);

    // head: out = h2bf @ Wm + bm
    dim3 g3(ceil_div_u(FO, BN), ceil_div_u(N, BM));
    k_gemm_bf16<<<g3, T, 0, stream>>>(h2bf, wmb, out, bm, N, FO, F2);
}
